// MultiheadEnoughAttention_63282048139392
// MI455X (gfx1250) — matrix operand order repaired and checked by reading, not compiled, not run on hardware
//
#include <hip/hip_runtime.h>
#include <math.h>

#define D_MODEL 512
#define NSEQ    8192
#define NBATCH  4

typedef __attribute__((ext_vector_type(16))) __bf16 bf16x16;
typedef __attribute__((ext_vector_type(2)))  __bf16 bf16x2;
typedef __attribute__((ext_vector_type(8)))  float  floatx8;

union FragAB {
  uint4   u[2];
  bf16x16 v;
};

// Native casts: clang emits v_cvt_pk_bf16_f32 (RNE) when the target has it.
__device__ __forceinline__ unsigned short f2bf(float f) {
  __bf16 h = (__bf16)f;
  return __builtin_bit_cast(unsigned short, h);
}
__device__ __forceinline__ unsigned int pack2bf(float a, float b) {
  bf16x2 t = { (__bf16)a, (__bf16)b };
  return __builtin_bit_cast(unsigned int, t);
}

constexpr int BM = 128, BN = 64, BK = 32, PK = 40;  // 80B LDS row pitch, 16B aligned

// ---- staging helpers: global->regs (issue only), regs->LDS (forces the wait) ----
template<bool TRANS_A>
__device__ __forceinline__ void load_a_regs(const float* __restrict__ Ab, int lda,
                                            int m0, int k0, int tid, float4 va[4]) {
  if (!TRANS_A) {
    const int am = tid >> 3, ak = (tid & 7) * 4;
    #pragma unroll
    for (int i = 0; i < 4; ++i)
      va[i] = *(const float4*)(Ab + (size_t)(m0 + am + i * 32) * lda + (k0 + ak));
  } else {
    const int ak = tid >> 5, am = (tid & 31) * 4;
    #pragma unroll
    for (int i = 0; i < 4; ++i)
      va[i] = *(const float4*)(Ab + (size_t)(k0 + ak + i * 8) * lda + (m0 + am));
  }
}

template<bool TRANS_B>
__device__ __forceinline__ void load_b_regs(const float* __restrict__ Bb, int ldb,
                                            int n0, int k0, int tid, float4 vb[2]) {
  if (!TRANS_B) {
    const int bk = tid >> 4, bn = (tid & 15) * 4;
    #pragma unroll
    for (int i = 0; i < 2; ++i)
      vb[i] = *(const float4*)(Bb + (size_t)(k0 + bk + i * 16) * ldb + (n0 + bn));
  } else {
    const int bn = tid >> 3, bk = (tid & 7) * 4;
    #pragma unroll
    for (int i = 0; i < 2; ++i)
      vb[i] = *(const float4*)(Bb + (size_t)(n0 + bn + i * 32) * ldb + (k0 + bk));
  }
}

template<bool TRANS_A>
__device__ __forceinline__ void store_a_lds(unsigned short (*__restrict__ As)[PK],
                                            int tid, const float4 va[4]) {
  if (!TRANS_A) {
    const int am = tid >> 3, ak = (tid & 7) * 4;
    #pragma unroll
    for (int i = 0; i < 4; ++i) {
      *(unsigned int*)&As[am + i * 32][ak]     = pack2bf(va[i].x, va[i].y);
      *(unsigned int*)&As[am + i * 32][ak + 2] = pack2bf(va[i].z, va[i].w);
    }
  } else {
    const int ak = tid >> 5, am = (tid & 31) * 4;
    #pragma unroll
    for (int i = 0; i < 4; ++i) {
      As[am + 0][ak + i * 8] = f2bf(va[i].x);
      As[am + 1][ak + i * 8] = f2bf(va[i].y);
      As[am + 2][ak + i * 8] = f2bf(va[i].z);
      As[am + 3][ak + i * 8] = f2bf(va[i].w);
    }
  }
}

template<bool TRANS_B>
__device__ __forceinline__ void store_b_lds(unsigned short (*__restrict__ Bs)[PK],
                                            int tid, const float4 vb[2]) {
  if (!TRANS_B) {
    const int bk = tid >> 4, bn = (tid & 15) * 4;
    #pragma unroll
    for (int i = 0; i < 2; ++i) {
      Bs[bn + 0][bk + i * 16] = f2bf(vb[i].x);
      Bs[bn + 1][bk + i * 16] = f2bf(vb[i].y);
      Bs[bn + 2][bk + i * 16] = f2bf(vb[i].z);
      Bs[bn + 3][bk + i * 16] = f2bf(vb[i].w);
    }
  } else {
    const int bn = tid >> 3, bk = (tid & 7) * 4;
    #pragma unroll
    for (int i = 0; i < 2; ++i) {
      *(unsigned int*)&Bs[bn + i * 32][bk]     = pack2bf(vb[i].x, vb[i].y);
      *(unsigned int*)&Bs[bn + i * 32][bk + 2] = pack2bf(vb[i].z, vb[i].w);
    }
  }
}

// ---------------------------------------------------------------------------
// C[bz] = alpha * op(A[bz]) @ op(B[bz]) (+ rowAdd[bz][n] broadcast per row)
//   op(A): TRANS_A ? A_mem[k,m] : A_mem[m,k]
//   op(B): TRANS_B ? B_mem[n,k] : B_mem[k,n]
// BM=128, BN=64, BK=32; 256 threads = 8 waves (4x2), wave tile 32x32.
// Double-buffered LDS software pipeline: global loads for tile k+1 issue
// before the barrier; their s_wait_loadcnt lands after the 4 WMMAs.
// M,N,K multiples of 128/64/32 (true here: 512 and 8192).
// ---------------------------------------------------------------------------
template<bool TRANS_A, bool TRANS_B>
__global__ __launch_bounds__(256)
void gemm_bf16_wmma(const float* __restrict__ A, size_t sA, int lda,
                    const float* __restrict__ B, size_t sB, int ldb,
                    float* __restrict__ C, size_t sC, int ldc,
                    const float* __restrict__ rowAdd, int sR,
                    int M, int N, int K, float alpha)
{
  (void)M; (void)N;
  __shared__ unsigned short As[2][BM][PK];   // As[buf][m][k], k-contiguous
  __shared__ unsigned short Bs[2][BN][PK];   // Bs[buf][n][k], k-contiguous

  const int tid  = threadIdx.x;
  const int wid  = tid >> 5;
  const int lane = tid & 31;
  const int lh   = lane >> 4;     // lane half: selects K sub-range (contiguous k order)
  const int lm   = lane & 15;
  const int wm   = wid >> 1;      // 0..3
  const int wn   = wid & 1;       // 0..1

  const int bz = blockIdx.z;
  const int m0 = blockIdx.y * BM;
  const int n0 = blockIdx.x * BN;

  const float* Ab = A + (size_t)bz * sA;
  const float* Bb = B + (size_t)bz * sB;
  float*       Cb = C + (size_t)bz * sC;

  floatx8 acc[2][2] = {};
  float4 va[4], vb[2];

  // prologue: stage tile 0 into buffer 0
  load_a_regs<TRANS_A>(Ab, lda, m0, 0, tid, va);
  load_b_regs<TRANS_B>(Bb, ldb, n0, 0, tid, vb);
  store_a_lds<TRANS_A>(As[0], tid, va);
  store_b_lds<TRANS_B>(Bs[0], tid, vb);

  int buf = 0;
  for (int k0 = 0; k0 < K; k0 += BK) {
    const bool has_next = (k0 + BK < K);
    if (has_next) {           // issue next tile's global loads (no wait yet)
      load_a_regs<TRANS_A>(Ab, lda, m0, k0 + BK, tid, va);
      load_b_regs<TRANS_B>(Bb, ldb, n0, k0 + BK, tid, vb);
    }
    __syncthreads();          // current buffer's LDS stores visible block-wide

    // ---- fragments in contiguous k order (k = 16*lh + s), 4 WMMAs per wave ----
    FragAB a[2], b[2];
    #pragma unroll
    for (int s = 0; s < 2; ++s) {
      // A 16x32 bf16: M=lm, K = lh*16 .. +15 (32 contiguous bytes)
      const uint4* pa = (const uint4*)&As[buf][wm * 32 + s * 16 + lm][lh * 16];
      a[s].u[0] = pa[0];
      a[s].u[1] = pa[1];      // +16B = K+8
      // B 32x16 bf16: N=lm, K = lh*16 .. +15 (32 contiguous bytes)
      const uint4* pb = (const uint4*)&Bs[buf][wn * 32 + s * 16 + lm][lh * 16];
      b[s].u[0] = pb[0];
      b[s].u[1] = pb[1];
    }
    #pragma unroll
    for (int sm = 0; sm < 2; ++sm)
      #pragma unroll
      for (int sn = 0; sn < 2; ++sn)
        acc[sm][sn] = __builtin_amdgcn_wmma_f32_16x16x32_bf16(
            false, a[sm].v, false, b[sn].v, (short)0, acc[sm][sn], false, false);

    if (has_next) {           // convert + store into the other buffer
      store_a_lds<TRANS_A>(As[buf ^ 1], tid, va);
      store_b_lds<TRANS_B>(Bs[buf ^ 1], tid, vb);
    }
    buf ^= 1;
  }

  // ---- epilogue: C/D layout m = r + 8*lh, n = lm ----
  #pragma unroll
  for (int sm = 0; sm < 2; ++sm) {
    #pragma unroll
    for (int sn = 0; sn < 2; ++sn) {
      const int gm0 = m0 + wm * 32 + sm * 16 + lh * 8;
      const int gn  = n0 + wn * 32 + sn * 16 + lm;
      const float addv = rowAdd ? rowAdd[(size_t)bz * sR + gn] : 0.0f;
      #pragma unroll
      for (int r = 0; r < 8; ++r)
        Cb[(size_t)(gm0 + r) * ldc + gn] = acc[sm][sn][r] * alpha + addv;
    }
  }
}

// xsum[b][d] = sum over n of X[b][n][d]   (coalesced, atomic-combined)
__global__ __launch_bounds__(512)
void colsum_kernel(const float* __restrict__ X, float* __restrict__ xsum)
{
  const int b = blockIdx.x;
  const int chunk = blockIdx.y;
  const int t = threadIdx.x;
  const float* Xb = X + ((size_t)b * NSEQ + (size_t)chunk * 256) * D_MODEL;
  float s = 0.0f;
  for (int r = 0; r < 256; ++r) s += Xb[(size_t)r * D_MODEL + t];
  atomicAdd(&xsum[b * D_MODEL + t], s);
}

// first[b][j] = (1/n) * sum_d xsum[b][d] * Wv[j][d]
__global__ __launch_bounds__(256)
void first_kernel(const float* __restrict__ xsum, const float* __restrict__ Wv,
                  float* __restrict__ first)
{
  const int b = blockIdx.x;
  const int j = blockIdx.y * 256 + threadIdx.x;
  const float* xs = xsum + b * D_MODEL;
  const float* w  = Wv + (size_t)j * D_MODEL;
  float s = 0.0f;
  for (int d = 0; d < D_MODEL; ++d) s += xs[d] * w[d];
  first[b * D_MODEL + j] = s * (1.0f / (float)NSEQ);
}

extern "C" void kernel_launch(void* const* d_in, const int* in_sizes, int n_in,
                              void* d_out, int out_size, void* d_ws, size_t ws_size,
                              hipStream_t stream)
{
  (void)in_sizes; (void)n_in; (void)out_size; (void)ws_size;
  const float* X  = (const float*)d_in[0];
  const float* Wv = (const float*)d_in[1];
  const float* Wt = (const float*)d_in[2];
  float* out = (float*)d_out;

  const int D = D_MODEL, n = NSEQ, B = NBATCH;
  const size_t dd = (size_t)D * D;

  // workspace layout (fp32): G | tmp | kv | xsum | first  (~12 MiB total)
  float* G     = (float*)d_ws;
  float* tmp   = G    + (size_t)B * dd;
  float* kv    = tmp  + (size_t)B * dd;
  float* xsum  = kv   + (size_t)B * dd;
  float* first = xsum + (size_t)B * D;

  hipMemsetAsync(xsum, 0, (size_t)B * D * sizeof(float), stream);
  colsum_kernel<<<dim3(B, n / 256), 512, 0, stream>>>(X, xsum);
  first_kernel<<<dim3(B, D / 256), 256, 0, stream>>>(xsum, Wv, first);

  // G[b] = X[b]^T @ X[b]        (M=N=512, K=8192)
  gemm_bf16_wmma<true, false><<<dim3(D / 64, D / 128, B), 256, 0, stream>>>(
      X, (size_t)n * D, D,  X, (size_t)n * D, D,  G, dd, D,
      nullptr, 0, D, D, n, 1.0f);

  // tmp[b] = W_theta @ G[b]     (512^3)
  gemm_bf16_wmma<false, false><<<dim3(D / 64, D / 128, B), 256, 0, stream>>>(
      Wt, 0, D,  G, dd, D,  tmp, dd, D,
      nullptr, 0, D, D, D, 1.0f);

  // kv[b] = alpha * tmp[b] @ W_v^T,  alpha = 1/(n*sqrt(d))
  const float alpha = 1.0f / ((float)n * sqrtf((float)D));
  gemm_bf16_wmma<false, true><<<dim3(D / 64, D / 128, B), 256, 0, stream>>>(
      tmp, dd, D,  Wv, 0, D,  kv, dd, D,
      nullptr, 0, D, D, D, alpha);

  // out[b] = X[b] @ kv[b] + first[b]   (M=8192, N=512, K=512)
  gemm_bf16_wmma<false, false><<<dim3(D / 64, n / 128, B), 256, 0, stream>>>(
      X, (size_t)n * D, D,  kv, dd, D,  out, (size_t)n * D, D,
      first, D, n, D, D, 1.0f);
}
